// FeedForward_82214263980557
// MI455X (gfx1250) — compile-verified, matrix operand order checked by reading, not run on hardware
//
#include <hip/hip_runtime.h>
#include <hip/hip_bf16.h>
#include <math.h>

typedef __bf16 bf16_t;
typedef __bf16 v16bf __attribute__((ext_vector_type(16)));
typedef __bf16 v8bf  __attribute__((ext_vector_type(8)));
typedef float  v8f   __attribute__((ext_vector_type(8)));
typedef int    v4i   __attribute__((ext_vector_type(4)));
typedef int    v8i   __attribute__((ext_vector_type(8)));
typedef unsigned int v4u __attribute__((ext_vector_type(4)));

#define BM  128
#define BN  128
#define BK  32
#define LDK 40   // padded LDS row stride in bf16 elements (80 B, 16B-aligned, bank-conflict-free)

// ---------------- CDNA5 data-mover selection: TDM > async-LDS > registers ----------------
#if defined(__gfx1250__) && __has_builtin(__builtin_amdgcn_tensor_load_to_lds)
#define USE_TDM 1
#else
#define USE_TDM 0
#endif
#if defined(__gfx1250__) && __has_builtin(__builtin_amdgcn_global_load_async_to_lds_b128)
#define USE_ASYNC_LDS 1
#else
#define USE_ASYNC_LDS 0
#endif

#if USE_TDM
// Issue one TDM 2D tile load: rows x cols = tile_d1 x tile_d0 (bf16 elements),
// row-major source with row stride = tensor_d0 elements, into LDS with
// padding 4 DWORDs after every 16 DWORDs (=> LDK=40 element row stride).
__device__ __forceinline__ void tdm_load_2d(unsigned lds_off, const void* gaddr,
                                            unsigned tensor_d0, unsigned tensor_d1,
                                            unsigned tile_d0, unsigned tile_d1) {
  const unsigned long long ga = (unsigned long long)(size_t)gaddr;
  v4u g0;
  g0[0] = 1u;                                            // count=1, user descriptor
  g0[1] = lds_off;                                       // LDS byte address
  g0[2] = (unsigned)(ga & 0xFFFFFFFFu);                  // global_addr[31:0]
  g0[3] = ((unsigned)(ga >> 32) & 0x01FFFFFFu)           // global_addr[56:32]
          | 0x80000000u;                                 // type=2 ("image")
  v8i g1;
  g1[0] = (int)0x06D10000u;  // data_size=1(2B) | pad_enable | pad_interval=3(16DW) | pad_amount=3(4DW)
  g1[1] = (int)((tensor_d0 & 0xFFFFu) << 16);            // abar_addr=0 | tensor_dim0[15:0]
  g1[2] = (int)((tensor_d0 >> 16) | ((tensor_d1 & 0xFFFFu) << 16)); // td0[31:16] | td1[15:0]
  g1[3] = (int)((tensor_d1 >> 16) | (tile_d0 << 16));    // td1[31:16] | tile_dim0
  g1[4] = (int)(tile_d1 & 0xFFFFu);                      // tile_dim1 | tile_dim2=0
  g1[5] = (int)tensor_d0;                                // tensor_dim0_stride[31:0] (elements)
  g1[6] = 0;                                             // stride[47:32] | dim1_stride[15:0]
  g1[7] = 0;
  const v4i z4 = {0, 0, 0, 0};
  const v8i z8 = {0, 0, 0, 0, 0, 0, 0, 0};
  __builtin_amdgcn_tensor_load_to_lds(g0, g1, z4, z4, z8, 0);
}
#define WAIT_TDM(n) __builtin_amdgcn_s_wait_tensorcnt(n)
#endif

#if USE_ASYNC_LDS
__device__ __forceinline__ void async_cp16(void* lds_dst, const void* gsrc) {
  __builtin_amdgcn_global_load_async_to_lds_b128(
      (__attribute__((address_space(1))) v4i*)gsrc,
      (__attribute__((address_space(3))) v4i*)lds_dst, 0, 0);
}
#if __has_builtin(__builtin_amdgcn_s_wait_asynccnt)
#define WAIT_ASYNC(n) __builtin_amdgcn_s_wait_asynccnt(n)
#else
#define WAIT_ASYNC(n)                                                     \
  do {                                                                    \
    if ((n) == 0) asm volatile("s_wait_asynccnt 0x0" ::: "memory");       \
    else          asm volatile("s_wait_asynccnt 0x4" ::: "memory");       \
  } while (0)
#endif
#endif

union FragU { v16bf v; v8bf h[2]; };
union Pack8 { v8bf  v; bf16_t e[8]; };

// Fragment load per CDNA5 ISA 7.12.2 16-bit A/B layout.
__device__ __forceinline__ v16bf frag_ld(const bf16_t* base, int row0, int lane) {
  const int sub = lane & 15;
  const int hi  = lane >> 4;
  const bf16_t* p = base + (row0 + sub) * LDK + (hi << 3);
  FragU f;
  f.h[0] = *(const v8bf*)(p);
  f.h[1] = *(const v8bf*)(p + 16);
  return f.v;
}

// Branch-free exact-GELU: erf via Abramowitz-Stegun 7.1.26 (|eps| < 1.5e-7).
__device__ __forceinline__ float gelu_exact(float x) {
  const float s = x * 0.70710678118654752f;
  const float a = fabsf(s);
  const float t = __builtin_amdgcn_rcpf(1.0f + 0.3275911f * a);
  const float y = t * (0.254829592f +
                  t * (-0.284496736f +
                  t * (1.421413741f +
                  t * (-1.453152027f +
                  t * 1.061405429f))));
  float erf = 1.0f - y * __expf(-a * a);
  erf = copysignf(erf, s);
  return 0.5f * x * (1.0f + erf);
}

// ---------------- dtype conversion pass (fp32/int32 -> bf16) ----------------
__global__ __launch_bounds__(256) void cvt_f32_to_bf16(
    const float* __restrict__ s, bf16_t* __restrict__ d, int n8) {
  const int i = blockIdx.x * 256 + threadIdx.x;
  if (i >= n8) return;
  const float4* p = (const float4*)s + (size_t)i * 2;
  const float4 f0 = p[0], f1 = p[1];
  Pack8 o;
  o.e[0] = (bf16_t)f0.x; o.e[1] = (bf16_t)f0.y; o.e[2] = (bf16_t)f0.z; o.e[3] = (bf16_t)f0.w;
  o.e[4] = (bf16_t)f1.x; o.e[5] = (bf16_t)f1.y; o.e[6] = (bf16_t)f1.z; o.e[7] = (bf16_t)f1.w;
  *((v8bf*)d + i) = o.v;
}

__global__ __launch_bounds__(256) void cvt_i32_to_bf16(
    const int* __restrict__ s, bf16_t* __restrict__ d, int n8) {
  const int i = blockIdx.x * 256 + threadIdx.x;
  if (i >= n8) return;
  const int4* p = (const int4*)s + (size_t)i * 2;
  const int4 q0 = p[0], q1 = p[1];
  Pack8 o;   // int8-range values are exact in bf16
  o.e[0] = (bf16_t)(float)q0.x; o.e[1] = (bf16_t)(float)q0.y;
  o.e[2] = (bf16_t)(float)q0.z; o.e[3] = (bf16_t)(float)q0.w;
  o.e[4] = (bf16_t)(float)q1.x; o.e[5] = (bf16_t)(float)q1.y;
  o.e[6] = (bf16_t)(float)q1.z; o.e[7] = (bf16_t)(float)q1.w;
  *((v8bf*)d + i) = o.v;
}

// ---------------- unified bf16 GEMM, WMMA core, TDM/async-LDS pipeline ----------------
// MODE 0: out_bf16 = gelu(A @ B^T + c0)          (up-proj + GELU)
// MODE 1: out_f32  = (A @ B^T) * c0 + c1          (down-proj dequant epilogue)
template <int MODE>
__global__ __launch_bounds__(256) void gemm_bf16(
    const bf16_t* __restrict__ A,    // [Mtot, K] row-major
    const bf16_t* __restrict__ Bm,   // [Ntot, K] row-major (K-contiguous)
    const float*  __restrict__ c0,
    const float*  __restrict__ c1,
    bf16_t* __restrict__ outb,
    float*  __restrict__ outf,
    int K, int Ntot, int Mtot)
{
  __shared__ __align__(16) bf16_t lA[2][BM * LDK];
  __shared__ __align__(16) bf16_t lB[2][BN * LDK];

  const int t    = threadIdx.x;
  const int lane = t & 31;
  const int wave = t >> 5;
  const int wm   = wave & 1;   // 2 waves across M (64 rows each)
  const int wn   = wave >> 1;  // 4 waves across N (32 cols each)
  const int m0   = blockIdx.y * BM;
  const int n0   = blockIdx.x * BN;

  v8f acc[4][2] = {};
  const int KT = K / BK;

#if USE_TDM
  // Wave 0 drives the Tensor Data Mover; TENSORcnt is per-wave and in-order,
  // so a depth-2 descriptor pipeline double-buffers the LDS tiles.
  const bf16_t* gAt = A  + (size_t)m0 * K;   // tile origin, rows m0..m0+127
  const bf16_t* gBt = Bm + (size_t)n0 * K;   // tile origin, rows n0..n0+127
  const unsigned offA[2] = {
      (unsigned)(size_t)(__attribute__((address_space(3))) void*)&lA[0][0],
      (unsigned)(size_t)(__attribute__((address_space(3))) void*)&lA[1][0]};
  const unsigned offB[2] = {
      (unsigned)(size_t)(__attribute__((address_space(3))) void*)&lB[0][0],
      (unsigned)(size_t)(__attribute__((address_space(3))) void*)&lB[1][0]};

  if (wave == 0) {
    tdm_load_2d(offA[0], gAt, (unsigned)K, (unsigned)(Mtot - m0), BK, BM);
    tdm_load_2d(offB[0], gBt, (unsigned)K, (unsigned)(Ntot - n0), BK, BN);
  }
  for (int kt = 0; kt < KT; ++kt) {
    if (wave == 0) {
      if (kt + 1 < KT) {
        const int nb = (kt + 1) & 1;
        tdm_load_2d(offA[nb], gAt + (size_t)(kt + 1) * BK, (unsigned)K,
                    (unsigned)(Mtot - m0), BK, BM);
        tdm_load_2d(offB[nb], gBt + (size_t)(kt + 1) * BK, (unsigned)K,
                    (unsigned)(Ntot - n0), BK, BN);
        WAIT_TDM(2);   // tile kt's two descriptors have completed
      } else {
        WAIT_TDM(0);
      }
    }
    __syncthreads();   // tile published to all waves

    const bf16_t* bufA = lA[kt & 1];
    const bf16_t* bufB = lB[kt & 1];
    v16bf bfr[2];
#pragma unroll
    for (int j = 0; j < 2; ++j) bfr[j] = frag_ld(bufB, wn * 32 + j * 16, lane);
#pragma unroll
    for (int i = 0; i < 4; ++i) {
      v16bf af = frag_ld(bufA, wm * 64 + i * 16, lane);
#pragma unroll
      for (int j = 0; j < 2; ++j)
        acc[i][j] = __builtin_amdgcn_wmma_f32_16x16x32_bf16(
            false, af, false, bfr[j], (short)0, acc[i][j], false, false);
    }
    __syncthreads();   // all readers done before buffer refill
  }
#else
  // staging: thread t owns 16 bf16 (32 B) of row srow at col scol
  const int srow = t >> 1;
  const int scol = (t & 1) << 4;
  const bf16_t* gA = A  + (size_t)(m0 + srow) * K + scol;
  const bf16_t* gB = Bm + (size_t)(n0 + srow) * K + scol;
  bf16_t* sA = &lA[0][srow * LDK + scol];
  bf16_t* sB = &lB[0][srow * LDK + scol];
#if USE_ASYNC_LDS
  {
    async_cp16(sA,     gA);  async_cp16(sA + 8, gA + 8);
    async_cp16(sB,     gB);  async_cp16(sB + 8, gB + 8);
  }
  for (int kt = 0; kt < KT; ++kt) {
    if (kt + 1 < KT) {
      const bf16_t* a = gA + (size_t)(kt + 1) * BK;
      const bf16_t* b = gB + (size_t)(kt + 1) * BK;
      const int nb = (kt + 1) & 1;
      bf16_t* da = sA + nb * (BM * LDK);
      bf16_t* db = sB + nb * (BN * LDK);
      async_cp16(da,     a);  async_cp16(da + 8, a + 8);
      async_cp16(db,     b);  async_cp16(db + 8, b + 8);
      WAIT_ASYNC(4);
    } else {
      WAIT_ASYNC(0);
    }
    __syncthreads();
    const bf16_t* bufA = lA[kt & 1];
    const bf16_t* bufB = lB[kt & 1];
    v16bf bfr[2];
#pragma unroll
    for (int j = 0; j < 2; ++j) bfr[j] = frag_ld(bufB, wn * 32 + j * 16, lane);
#pragma unroll
    for (int i = 0; i < 4; ++i) {
      v16bf af = frag_ld(bufA, wm * 64 + i * 16, lane);
#pragma unroll
      for (int j = 0; j < 2; ++j)
        acc[i][j] = __builtin_amdgcn_wmma_f32_16x16x32_bf16(
            false, af, false, bfr[j], (short)0, acc[i][j], false, false);
    }
    __syncthreads();
  }
#else
  for (int kt = 0; kt < KT; ++kt) {
    v8bf ra0 = *(const v8bf*)(gA);
    v8bf ra1 = *(const v8bf*)(gA + 8);
    v8bf rb0 = *(const v8bf*)(gB);
    v8bf rb1 = *(const v8bf*)(gB + 8);
    gA += BK; gB += BK;
    __syncthreads();
    *(v8bf*)(sA) = ra0;  *(v8bf*)(sA + 8) = ra1;
    *(v8bf*)(sB) = rb0;  *(v8bf*)(sB + 8) = rb1;
    __syncthreads();
    v16bf bfr[2];
#pragma unroll
    for (int j = 0; j < 2; ++j) bfr[j] = frag_ld(lB[0], wn * 32 + j * 16, lane);
#pragma unroll
    for (int i = 0; i < 4; ++i) {
      v16bf af = frag_ld(lA[0], wm * 64 + i * 16, lane);
#pragma unroll
      for (int j = 0; j < 2; ++j)
        acc[i][j] = __builtin_amdgcn_wmma_f32_16x16x32_bf16(
            false, af, false, bfr[j], (short)0, acc[i][j], false, false);
    }
  }
#endif
#endif

  // Epilogue. C/D layout: vgpr r, lane L<16 -> (M=r,N=L); L>=16 -> (M=r+8,N=L-16)
  const int sub = lane & 15;
  const int hi  = lane >> 4;
#pragma unroll
  for (int j = 0; j < 2; ++j) {
    const int col = n0 + wn * 32 + j * 16 + sub;
    const float k0 = c0[col];
    const float k1 = (MODE == 1) ? c1[col] : 0.0f;
#pragma unroll
    for (int i = 0; i < 4; ++i) {
      const int rbase = m0 + wm * 64 + i * 16 + hi * 8;
#pragma unroll
      for (int r = 0; r < 8; ++r) {
        const float v = acc[i][j][r];
        if (MODE == 0) {
          outb[(size_t)(rbase + r) * Ntot + col] = (bf16_t)gelu_exact(v + k0);
        } else {
          outf[(size_t)(rbase + r) * Ntot + col] = v * k0 + k1;
        }
      }
    }
  }
}

extern "C" void kernel_launch(void* const* d_in, const int* in_sizes, int n_in,
                              void* d_out, int out_size, void* d_ws, size_t ws_size,
                              hipStream_t stream) {
  (void)in_sizes; (void)n_in; (void)out_size; (void)ws_size;
  const float* x            = (const float*)d_in[0];
  const float* w_up         = (const float*)d_in[1];
  const float* b_up         = (const float*)d_in[2];
  const int*   w_down_q     = (const int*)  d_in[3];
  const float* w_down_scale = (const float*)d_in[4];
  const float* b_down       = (const float*)d_in[5];
  float* out = (float*)d_out;

  const size_t Btok = 4, S = 4096, D = 2048, DI = 8192;
  const size_t M = Btok * S;

  // workspace layout (bf16): h[M,DI] | x_bf[M,D] | wup_bf[DI,D] | wdn_bf[D,DI]
  bf16_t* hbuf   = (bf16_t*)d_ws;
  bf16_t* x_bf   = hbuf   + M * DI;
  bf16_t* wup_bf = x_bf   + M * D;
  bf16_t* wdn_bf = wup_bf + DI * D;

  dim3 blk(256);
  {
    const int n8 = (int)(M * D / 8);
    cvt_f32_to_bf16<<<dim3((n8 + 255) / 256), blk, 0, stream>>>(x, x_bf, n8);
  }
  {
    const int n8 = (int)(DI * D / 8);
    cvt_f32_to_bf16<<<dim3((n8 + 255) / 256), blk, 0, stream>>>(w_up, wup_bf, n8);
  }
  {
    const int n8 = (int)(D * DI / 8);
    cvt_i32_to_bf16<<<dim3((n8 + 255) / 256), blk, 0, stream>>>(w_down_q, wdn_bf, n8);
  }

  dim3 g1(DI / BN, M / BM);   // 64 x 128
  gemm_bf16<0><<<g1, blk, 0, stream>>>(x_bf, wup_bf, b_up, nullptr,
                                       hbuf, nullptr, (int)D, (int)DI, (int)M);

  dim3 g2(D / BN, M / BM);    // 16 x 128
  gemm_bf16<1><<<g2, blk, 0, stream>>>(hbuf, wdn_bf, w_down_scale, b_down,
                                       nullptr, out, (int)DI, (int)D, (int)M);
}
